// Attention_14310831030361
// MI455X (gfx1250) — matrix operand order repaired and checked by reading, not compiled, not run on hardware
//
#include <hip/hip_runtime.h>
#include <hip/hip_bf16.h>

// Transformer-XL relative attention for MI455X (gfx1250, wave32, WMMA).
// Pipeline:
//   1) pack_kernel : q+u, q+v, K -> bf16 [b][n][i][d]; V -> bf16 transposed [b][n][d][j]
//   2) rh_kernel   : rh = r @ Wr^T + br via v_wmma_f32_16x16x32_bf16, stored bf16 [n][k][d]
//   3) attn_kernel : per (b, n, 32-row q tile): AC GEMM + rel-shifted BD GEMM scatter-add
//                    into a 32x1024 f32 LDS score block, mask+softmax in LDS,
//                    weights spilled bf16 for the head-mean, PV GEMM -> out
//   4) reduce_mean : deterministic mean over the 16 heads -> second output region
// All fragment loads are contiguous 16B chunks per lane -> b128 vector loads.

#define L_SEQ 1024
#define BATCH 4
#define NH    16
#define DH    64
#define MQ    32

typedef __attribute__((ext_vector_type(16))) __bf16        v16bf;
typedef __attribute__((ext_vector_type(8)))  float         v8f;
typedef __attribute__((ext_vector_type(4)))  float         v4f;
typedef __attribute__((ext_vector_type(4)))  unsigned int  v4u;

union FragU  { v16bf v; v4u u[2]; unsigned short s[16]; };

__device__ inline unsigned short f2bf(float f) {
  unsigned int x = __float_as_uint(f);
  x += 0x7FFFu + ((x >> 16) & 1u);        // round-to-nearest-even
  return (unsigned short)(x >> 16);
}
__device__ inline float bf2f(unsigned short h) {
  return __uint_as_float(((unsigned int)h) << 16);
}

// ---- Fragment loaders -------------------------------------------------------
// A fragment 16x32 (MxK), layout: lane m<16 (hi half +16 on K):
//   s[0..15] = row[k0+h16 .. +15]  (two 16B chunks)
__device__ inline v16bf ldA_bf(const unsigned short* base, int ld, int m0, int k0,
                               int lane) {
  int hlf = lane >> 4, mr = lane & 15;
  const unsigned short* row = base + (size_t)(m0 + mr) * ld + k0 + hlf * 16;
  FragU a;
  a.u[0] = *(const v4u*)(row);
  a.u[1] = *(const v4u*)(row + 8);
  return a.v;
}
__device__ inline v16bf ldA_f32(const float* base, int ld, int m0, int k0, int lane) {
  int hlf = lane >> 4, mr = lane & 15;
  const float* row = base + (size_t)(m0 + mr) * ld + k0 + hlf * 16;
  v4f f0 = *(const v4f*)(row);
  v4f f1 = *(const v4f*)(row + 4);
  v4f f2 = *(const v4f*)(row + 8);
  v4f f3 = *(const v4f*)(row + 12);
  FragU a;
#pragma unroll
  for (int p = 0; p < 4; ++p) {
    a.s[p]      = f2bf(f0[p]);
    a.s[p + 4]  = f2bf(f1[p]);
    a.s[p + 8]  = f2bf(f2[p]);
    a.s[p + 12] = f2bf(f3[p]);
  }
  return a.v;
}
// B fragment 32x16 (KxN), source row-major [N][ld] (row n holds contiguous K):
//   lane n<16: K = k0+h16 .. +15  (one 32B chunk)
__device__ inline v16bf ldB_bf(const unsigned short* base, int ld, int n0, int k0,
                               int lane) {
  int hlf = lane >> 4, nr = lane & 15;
  const unsigned short* row = base + (size_t)(n0 + nr) * ld + k0 + hlf * 16;
  FragU b;
  b.u[0] = *(const v4u*)(row);
  b.u[1] = *(const v4u*)(row + 8);
  return b.v;
}
__device__ inline v16bf ldB_f32(const float* base, int ld, int n0, int k0, int lane) {
  int hlf = lane >> 4, nr = lane & 15;
  const float* row = base + (size_t)(n0 + nr) * ld + k0 + hlf * 16;
  FragU b;
#pragma unroll
  for (int c = 0; c < 4; ++c) {
    v4f f = *(const v4f*)(row + 4 * c);
#pragma unroll
    for (int p = 0; p < 4; ++p) b.s[4 * c + p] = f2bf(f[p]);
  }
  return b.v;
}

__device__ inline v8f wmma_bf16(v16bf a, v16bf b, v8f c) {
  return __builtin_amdgcn_wmma_f32_16x16x32_bf16(false, a, false, b, (short)0, c,
                                                 false, false);
}

// ---------------------------------------------------------------------------
__global__ void pack_kernel(const float* __restrict__ q, const float* __restrict__ Kin,
                            const float* __restrict__ Vin, const float* __restrict__ u,
                            const float* __restrict__ vv,
                            unsigned short* __restrict__ QU, unsigned short* __restrict__ QV,
                            unsigned short* __restrict__ KB, unsigned short* __restrict__ VT) {
  int idx = blockIdx.x * blockDim.x + threadIdx.x;
  if (idx >= BATCH * NH * L_SEQ * DH) return;
  int d = idx & 63;
  int i = (idx >> 6) & 1023;
  int n = (idx >> 16) & 15;
  int b = idx >> 20;
  int c = n * DH + d;
  float qval = q[((size_t)i * BATCH + b) * 1024 + c];
  QU[idx] = f2bf(qval + u[c]);
  QV[idx] = f2bf(qval + vv[c]);
  KB[idx] = f2bf(Kin[((size_t)i * BATCH + b) * 1024 + c]);
  // VT: [b][n][d2][j]  (same b,n bits; j innermost for contiguous K-dim reads in PV)
  int j  = idx & 1023;
  int d2 = (idx >> 10) & 63;
  VT[idx] = f2bf(Vin[((size_t)j * BATCH + b) * 1024 + n * DH + d2]);
}

// ---------------------------------------------------------------------------
// rh[k][c] = sum_p r[k][p] * Wr[c][p] + br[c], stored bf16 as RB[n][k][d], c = n*64+d
__global__ void rh_kernel(const float* __restrict__ r, const float* __restrict__ Wr,
                          const float* __restrict__ br, unsigned short* __restrict__ RB) {
  int lane = threadIdx.x;
  int kt = blockIdx.x * 16;   // output row tile (position index)
  int ct = blockIdx.y * 16;   // output col tile (d_model index)
  v8f acc = {};
  for (int p0 = 0; p0 < 1024; p0 += 32) {
    v16bf a  = ldA_f32(r  + (size_t)kt * 1024, 1024, 0, p0, lane);
    v16bf bm = ldB_f32(Wr + (size_t)ct * 1024, 1024, 0, p0, lane);
    acc = wmma_bf16(a, bm, acc);
  }
  int hlf = lane >> 4, nn = lane & 15;
  int c = ct + nn;
  float bias = br[c];
  int nh = c >> 6, d = c & 63;
#pragma unroll
  for (int e = 0; e < 8; ++e) {
    int k = kt + e + hlf * 8;
    RB[((size_t)nh * L_SEQ + k) * DH + d] = f2bf(acc[e] + bias);
  }
}

// ---------------------------------------------------------------------------
__global__ __launch_bounds__(256) void attn_kernel(
    const unsigned short* __restrict__ QU, const unsigned short* __restrict__ QV,
    const unsigned short* __restrict__ KB, const unsigned short* __restrict__ VT,
    const unsigned short* __restrict__ RB, const unsigned char* __restrict__ mask,
    unsigned short* __restrict__ WW, float* __restrict__ out) {
  extern __shared__ char smem_raw[];
  float* S      = (float*)smem_raw;       // MQ * L_SEQ score/weight block (128 KB)
  float* red    = S + MQ * L_SEQ;         // MQ * 8 reduction scratch
  float* rowmax = red + MQ * 8;           // MQ
  float* rowsum = rowmax + MQ;            // MQ
  unsigned char* maskS = (unsigned char*)(rowsum + MQ);  // L_SEQ bytes

  const int i0   = blockIdx.x * MQ;
  const int b    = blockIdx.y;
  const int n    = blockIdx.z;
  const int tid  = threadIdx.x;
  const int wid  = tid >> 5, lane = tid & 31;
  const int hlf  = lane >> 4, lidx = lane & 15;

  const size_t bn = (size_t)(b * NH + n);
  const unsigned short* QUb = QU + (bn * L_SEQ + i0) * DH;
  const unsigned short* QVb = QV + (bn * L_SEQ + i0) * DH;
  const unsigned short* KBb = KB + bn * L_SEQ * DH;
  const unsigned short* VTb = VT + bn * (size_t)DH * L_SEQ;
  const unsigned short* RBb = RB + (size_t)n * L_SEQ * DH;
  unsigned short* WWb = WW + bn * (size_t)L_SEQ * L_SEQ;

  // Stage mask bytes once (reused 128x per thread in the softmax passes).
  for (int j = tid; j < L_SEQ; j += 256) maskS[j] = mask[j * BATCH + b];

  // ---- AC = (q+u) @ K^T : 2 x 64 output tiles of 16x16, 2 WMMAs each ----
  for (int t = wid; t < (MQ / 16) * (L_SEQ / 16); t += 8) {
    int tm = t >> 6;
    int tj = t & 63;
    v8f acc = {};
#pragma unroll
    for (int ks = 0; ks < DH; ks += 32) {
      v16bf a  = ldA_bf(QUb, DH, tm * 16, ks, lane);
      v16bf bm = ldB_bf(KBb, DH, tj * 16, ks, lane);
      acc = wmma_bf16(a, bm, acc);
    }
#pragma unroll
    for (int e = 0; e < 8; ++e)
      S[(tm * 16 + e + hlf * 8) * L_SEQ + tj * 16 + lidx] = acc[e];
  }
  __syncthreads();

  // ---- Braw = (q+v) @ rh^T over rows [i0, i0+MQ], rel_shift scatter-add into S.
  // rel_shift: BD[i,j] = Braw[i, j-i+L-1] (j<=i); 0 (j==i+1); Braw[i+1, j-i-2] (j>=i+2).
  // Each Braw element targets exactly one BD element -> no atomics needed.
  for (int t = wid; t < (MQ / 16 + 1) * (L_SEQ / 16); t += 8) {
    int tr = t >> 6;
    int tk = t & 63;
    v8f acc = {};
#pragma unroll
    for (int ks = 0; ks < DH; ks += 32) {
      v16bf a  = ldA_bf(QVb, DH, tr * 16, ks, lane);  // last block reads past QV: garbage rows dropped below
      v16bf bm = ldB_bf(RBb, DH, tk * 16, ks, lane);
      acc = wmma_bf16(a, bm, acc);
    }
    int kg = tk * 16 + lidx;
#pragma unroll
    for (int e = 0; e < 8; ++e) {
      int rl = tr * 16 + e + hlf * 8;  // local Braw row
      int rg = i0 + rl;                // global Braw row
      if (kg + rg >= L_SEQ - 1) {
        if (rl < MQ) S[rl * L_SEQ + (kg + rg - (L_SEQ - 1))] += acc[e];
      } else {
        int il = rl - 1;
        if (il >= 0 && il < MQ) S[il * L_SEQ + (kg + rg + 1)] += acc[e];
      }
    }
  }
  __syncthreads();

  // ---- scale + mask + softmax over j (8 threads per row) ----
  {
    int rr = tid >> 3, l8 = tid & 7;
    float mx = -3.0e38f;
    for (int j = l8; j < L_SEQ; j += 8) {
      float sc = S[rr * L_SEQ + j] * 0.125f;          // / sqrt(64)
      if (maskS[j]) sc = -1.0e9f;                     // mask (1,klen,B,1), bool as byte
      S[rr * L_SEQ + j] = sc;
      mx = fmaxf(mx, sc);
    }
    red[rr * 8 + l8] = mx;
  }
  __syncthreads();
  if (tid < MQ) {
    float m = red[tid * 8];
#pragma unroll
    for (int k = 1; k < 8; ++k) m = fmaxf(m, red[tid * 8 + k]);
    rowmax[tid] = m;
  }
  __syncthreads();
  {
    int rr = tid >> 3, l8 = tid & 7;
    float rm = rowmax[rr], sum = 0.f;
    for (int j = l8; j < L_SEQ; j += 8) {
      float ex = __expf(S[rr * L_SEQ + j] - rm);
      S[rr * L_SEQ + j] = ex;
      sum += ex;
    }
    red[rr * 8 + l8] = sum;
  }
  __syncthreads();
  if (tid < MQ) {
    float s = 0.f;
#pragma unroll
    for (int k = 0; k < 8; ++k) s += red[tid * 8 + k];
    rowsum[tid] = s;
  }
  __syncthreads();
  {
    int rr = tid >> 3, l8 = tid & 7;
    float inv = 1.0f / rowsum[rr];
    for (int j = l8; j < L_SEQ; j += 8) {
      float w = S[rr * L_SEQ + j] * inv;
      S[rr * L_SEQ + j] = w;
      WWb[(size_t)(i0 + rr) * L_SEQ + j] = f2bf(w);   // spill for head-mean reduction
    }
  }
  __syncthreads();

  // ---- out = W @ V : 8 output tiles (2 x 4), one per wave, 32 WMMAs each ----
  {
    int tm = wid >> 2, td = wid & 3;
    v8f acc = {};
    for (int ks = 0; ks < L_SEQ; ks += 32) {
      v16bf a  = ldA_f32(S + (size_t)(tm * 16) * L_SEQ, L_SEQ, 0, ks, lane);
      v16bf bm = ldB_bf(VTb, L_SEQ, td * 16, ks, lane);
      acc = wmma_bf16(a, bm, acc);
    }
#pragma unroll
    for (int e = 0; e < 8; ++e) {
      int i = i0 + tm * 16 + e + hlf * 8;
      out[((size_t)i * BATCH + b) * 1024 + n * DH + td * 16 + lidx] = acc[e];
    }
  }
}

// ---------------------------------------------------------------------------
// Mean over heads: each thread handles 8 consecutive j (one b128 load per head).
__global__ void reduce_mean_kernel(const unsigned short* __restrict__ WW,
                                   float* __restrict__ out2) {
  int idx = blockIdx.x * blockDim.x + threadIdx.x;
  if (idx >= BATCH * L_SEQ * (L_SEQ / 8)) return;
  int j8 = (idx & 127) * 8;
  int i  = (idx >> 7) & 1023;
  int b  = idx >> 17;
  float s[8] = {0.f, 0.f, 0.f, 0.f, 0.f, 0.f, 0.f, 0.f};
#pragma unroll
  for (int n = 0; n < NH; ++n) {
    const unsigned short* p =
        WW + (((size_t)(b * NH + n)) * L_SEQ + i) * L_SEQ + j8;
    v4u u = *(const v4u*)p;
#pragma unroll
    for (int e = 0; e < 8; ++e)
      s[e] += bf2f((unsigned short)((u[e >> 1] >> ((e & 1) * 16)) & 0xFFFFu));
  }
#pragma unroll
  for (int e = 0; e < 8; ++e)
    out2[((size_t)i * L_SEQ + j8 + e) * BATCH + b] = s[e] * (1.0f / NH);
}

// ---------------------------------------------------------------------------
extern "C" void kernel_launch(void* const* d_in, const int* in_sizes, int n_in,
                              void* d_out, int out_size, void* d_ws, size_t ws_size,
                              hipStream_t stream) {
  (void)in_sizes; (void)n_in; (void)out_size; (void)ws_size;
  const float* q  = (const float*)d_in[0];
  const float* K  = (const float*)d_in[1];
  const float* V  = (const float*)d_in[2];
  const unsigned char* mask = (const unsigned char*)d_in[3];  // jax bool -> byte
  const float* r  = (const float*)d_in[4];
  const float* u  = (const float*)d_in[5];
  const float* vv = (const float*)d_in[6];
  const float* Wr = (const float*)d_in[7];
  const float* br = (const float*)d_in[8];
  float* out = (float*)d_out;

  const size_t PK = (size_t)BATCH * NH * L_SEQ * DH;  // 4 Mi elements
  unsigned short* QU = (unsigned short*)d_ws;
  unsigned short* QV = QU + PK;
  unsigned short* KB = QV + PK;
  unsigned short* VT = KB + PK;
  unsigned short* RB = VT + PK;                        // NH*L*DH = 1 Mi
  unsigned short* WW = RB + (size_t)NH * L_SEQ * DH;   // B*NH*L*L = 64 Mi

  int tot = BATCH * NH * L_SEQ * DH;
  pack_kernel<<<(tot + 255) / 256, 256, 0, stream>>>(q, K, V, u, vv, QU, QV, KB, VT);
  rh_kernel<<<dim3(L_SEQ / 16, 1024 / 16), 32, 0, stream>>>(r, Wr, br, RB);

  size_t smem = (size_t)MQ * L_SEQ * 4 + (size_t)MQ * 8 * 4 + 2 * (size_t)MQ * 4
              + (size_t)L_SEQ;  // +1KB staged mask
  attn_kernel<<<dim3(L_SEQ / MQ, BATCH, NH), 256, smem, stream>>>(QU, QV, KB, VT, RB,
                                                                  mask, WW, out);
  int tot2 = BATCH * L_SEQ * (L_SEQ / 8);
  reduce_mean_kernel<<<(tot2 + 255) / 256, 256, 0, stream>>>(
      WW, out + (size_t)L_SEQ * BATCH * 1024);
}
